// CausalSelfAttention_71116068487587
// MI455X (gfx1250) — matrix operand order repaired and checked by reading, not compiled, not run on hardware
//
#include <hip/hip_runtime.h>
#include <cstdint>
#include <cstddef>

// Problem constants (match reference)
#define B_    2
#define T_    2048
#define C_    1024
#define NH_   16
#define HD_   64
#define WIN_  256
#define SINK_ 4

typedef __attribute__((ext_vector_type(16))) __bf16        v16bf;
typedef __attribute__((ext_vector_type(8)))  float         v8f;
typedef __attribute__((ext_vector_type(4)))  unsigned int  uint4v;
typedef __attribute__((ext_vector_type(2)))  unsigned int  uint2v;
typedef __attribute__((ext_vector_type(4)))  float         f4v;

// f32 -> bf16 bits using the native hardware convert (v_cvt_*_bf16_f32)
__device__ __forceinline__ unsigned short bf_bits(float f) {
    __bf16 h = (__bf16)f;
    return __builtin_bit_cast(unsigned short, h);
}

union Frag {
    v16bf          v;
    unsigned short u[16];
    uint4v         q[2];
};

// Async global->LDS copy of 16 bytes (GLOBAL_LOAD_ASYNC_TO_LDS_B128, GV mode).
// lds_off: wave-relative LDS byte address; gaddr: 64-bit global address.
__device__ __forceinline__ void async_ld_b128(unsigned lds_off, const void* gaddr) {
    asm volatile("global_load_async_to_lds_b128 %0, %1, off"
                 :: "v"(lds_off), "v"(gaddr) : "memory");
}

// ---------------------------------------------------------------------------
// Elementwise f32 -> bf16 (vectorized: float4 in, 4 packed bf16 out)
// ---------------------------------------------------------------------------
__global__ __launch_bounds__(256) void cvt_bf16(const float* __restrict__ src,
                                                unsigned short* __restrict__ dst,
                                                int n) {
    int i = (blockIdx.x * 256 + threadIdx.x) * 4;
    if (i < n) {
        f4v f = *(const f4v*)(src + i);
        union { unsigned short h[4]; uint2v u; } p;
        p.h[0] = bf_bits(f[0]);
        p.h[1] = bf_bits(f[1]);
        p.h[2] = bf_bits(f[2]);
        p.h[3] = bf_bits(f[3]);
        *(uint2v*)(dst + i) = p.u;
    }
}

// ---------------------------------------------------------------------------
// GEMM: C[M,N](f32) = A[M,K](bf16) * B[K,N](bf16), bf16 WMMA, f32 accumulate.
// M fixed at 4096 via grid.y*128. Tile: BM=128, BN=128, BK=32.
// 256 threads = 8 waves, wave grid 4(M) x 2(N), each wave 32x64 (2x4 WMMA accums).
// Double-buffered LDS: tile k+1 staged (A via GLOBAL_LOAD_ASYNC_TO_LDS_B128,
// B via VGPR transpose) while tile k runs on the WMMA pipe.
// ---------------------------------------------------------------------------
__global__ __launch_bounds__(256) void gemm_bf16(const unsigned short* __restrict__ A,
                                                 const unsigned short* __restrict__ Bw,
                                                 float* __restrict__ Cm,
                                                 int K, int N) {
    __shared__ __align__(16) unsigned short As[2][128][32];   // [buf][m][k]
    __shared__ __align__(16) unsigned short Bs[2][128][32];   // [buf][n][k]

    const int tid  = threadIdx.x;
    const int lane = tid & 31;
    const int wid  = tid >> 5;
    const int wm   = wid & 3;        // 0..3  (M waves)
    const int wn   = wid >> 2;       // 0..1  (N waves)
    const int m0   = blockIdx.y * 128;
    const int n0   = blockIdx.x * 128;
    const int half = lane >> 4;      // 0/1
    const int l15  = lane & 15;

    v8f acc[2][4];
    #pragma unroll
    for (int mi = 0; mi < 2; ++mi)
        #pragma unroll
        for (int ni = 0; ni < 4; ++ni)
            #pragma unroll
            for (int r = 0; r < 8; ++r) acc[mi][ni][r] = 0.0f;

    const int arow  = tid >> 1;            // 0..127
    const int akoff = (tid & 1) * 16;      // 0/16 (bf16 elements)
    const int bk    = tid >> 3;            // 0..31
    const int bnoff = (tid & 7) * 16;      // 0..112

    const unsigned as_off0 = (unsigned)(uintptr_t)(&As[0][arow][akoff]);
    const unsigned as_off1 = (unsigned)(uintptr_t)(&As[1][arow][akoff]);

    const unsigned short* ga = A  + (size_t)(m0 + arow) * K + akoff;  // += 32/iter
    const unsigned short* gb = Bw + (size_t)bk * N + n0 + bnoff;      // += 32N/iter
    const size_t gb_step = (size_t)32 * N;

    // ---- Prologue: stage tile 0 into buffer 0 ----
    async_ld_b128(as_off0,      ga);
    async_ld_b128(as_off0 + 16, ga + 8);
    {
        Frag tb;
        tb.q[0] = *(const uint4v*)gb;
        tb.q[1] = *(const uint4v*)(gb + 8);
        #pragma unroll
        for (int i = 0; i < 16; ++i) Bs[0][bnoff + i][bk] = tb.u[i];
    }

    const int niter = K >> 5;
    for (int it = 0; it < niter; ++it) {
        const int buf = it & 1;
        // Tile `it`'s A async copies (issued last iteration / prologue) landed:
        asm volatile("s_wait_asynccnt 0x0" ::: "memory");
        __syncthreads();   // all waves: A(it) in LDS, B(it) ds-stores visible

        // ---- Stage tile it+1 into the other buffer (overlaps with WMMA below)
        if (it + 1 < niter) {
            const unsigned short* ga_n = ga + (size_t)(it + 1) * 32;
            const unsigned short* gb_n = gb + (size_t)(it + 1) * gb_step;
            const unsigned aoff = buf ? as_off0 : as_off1;
            async_ld_b128(aoff,      ga_n);
            async_ld_b128(aoff + 16, ga_n + 8);
            Frag tb;
            tb.q[0] = *(const uint4v*)gb_n;
            tb.q[1] = *(const uint4v*)(gb_n + 8);
            #pragma unroll
            for (int i = 0; i < 16; ++i) Bs[buf ^ 1][bnoff + i][bk] = tb.u[i];
            __builtin_prefetch(ga_n + 32, 0, 1);            // global_prefetch_b8
        }

        // ---- Compute tile `it` ----
        Frag af[2], bf4[4];
        const int kb8 = half * 8;
        #pragma unroll
        for (int mi = 0; mi < 2; ++mi) {
            int r = wm * 32 + mi * 16 + l15;
            af[mi].q[0] = *(const uint4v*)&As[buf][r][kb8];
            af[mi].q[1] = *(const uint4v*)&As[buf][r][kb8 + 16];
        }
        const int kb16 = half * 8;
        #pragma unroll
        for (int ni = 0; ni < 4; ++ni) {
            int c = wn * 64 + ni * 16 + l15;
            bf4[ni].q[0] = *(const uint4v*)&Bs[buf][c][kb16];
            bf4[ni].q[1] = *(const uint4v*)&Bs[buf][c][kb16 + 16];
        }
        #pragma unroll
        for (int mi = 0; mi < 2; ++mi)
            #pragma unroll
            for (int ni = 0; ni < 4; ++ni)
                acc[mi][ni] = __builtin_amdgcn_wmma_f32_16x16x32_bf16(
                    false, af[mi].v, false, bf4[ni].v, (short)0, acc[mi][ni],
                    false, false);
    }

    // Epilogue: C layout — lanes 0-15 rows 0..7, lanes 16-31 rows 8..15, col = l15
    #pragma unroll
    for (int mi = 0; mi < 2; ++mi)
        #pragma unroll
        for (int ni = 0; ni < 4; ++ni)
            #pragma unroll
            for (int r = 0; r < 8; ++r) {
                int grow = m0 + wm * 32 + mi * 16 + r + 8 * half;
                int gcol = n0 + wn * 64 + ni * 16 + l15;
                Cm[(size_t)grow * N + gcol] = acc[mi][ni][r];
            }
}

// ---------------------------------------------------------------------------
// RoPE + split: qkv f32 [B,T,3C] -> q,k (rotated), v as bf16 [B*NH, T, HD]
// ---------------------------------------------------------------------------
__global__ __launch_bounds__(256) void rope_split(const float* __restrict__ qkv,
                                                  unsigned short* __restrict__ qb,
                                                  unsigned short* __restrict__ kb,
                                                  unsigned short* __restrict__ vb) {
    unsigned int i = blockIdx.x * 256u + threadIdx.x;   // 4,194,304 total
    int d = i & 63;
    int t = (i >> 6) & 2047;
    int h = (i >> 17) & 15;
    int b = i >> 21;
    size_t base = ((size_t)(b * T_ + t)) * (3 * C_) + h * HD_;
    int dd = d & 31;
    // inv_freq = 10000^(-2*dd/64) = exp(-ln(1e4)/32 * dd)
    float ang = (float)t * __expf(-0.2878231366242557f * (float)dd);
    float sn, cs;
    __sincosf(ang, &sn, &cs);
    int   partner = (d < 32) ? d + 32 : d - 32;
    float sign    = (d < 32) ? -1.0f : 1.0f;
    float qv = qkv[base + d],          qp = qkv[base + partner];
    float kv = qkv[base + C_ + d],     kp = qkv[base + C_ + partner];
    float vv = qkv[base + 2 * C_ + d];
    size_t ob = ((size_t)((b * NH_ + h) * T_ + t)) * HD_ + d;
    qb[ob] = bf_bits(qv * cs + sign * qp * sn);
    kb[ob] = bf_bits(kv * cs + sign * kp * sn);
    vb[ob] = bf_bits(vv);
}

// ---------------------------------------------------------------------------
// Flash-style sliding-window attention. One wave per 16-query tile.
// ---------------------------------------------------------------------------
__device__ __forceinline__ void attn_tile(int lane, int bh, int qt, int kt,
                                          const Frag* qf,
                                          const unsigned short* __restrict__ kbp,
                                          const unsigned short* __restrict__ vbp,
                                          v8f* o, float* mrow, float* lrow,
                                          unsigned short (*PsW)[16],
                                          unsigned short (*VsW)[32]) {
    const int half = lane >> 4;
    const int l15  = lane & 15;

    // Stage V key-tile into LDS transposed: Vs[d][k] (k=16..31 pre-zeroed)
    {
        int krow = lane >> 1, dpart = (lane & 1) * 32;
        const unsigned short* gv =
            vbp + ((size_t)(bh * T_ + kt * 16 + krow)) * HD_ + dpart;
        #pragma unroll
        for (int i = 0; i < 32; ++i) VsW[dpart + i][krow] = gv[i];
    }

    // S = Q * K^T : B-fragment = K^T (d contraction rows, key cols), contiguous in d
    v8f s;
    #pragma unroll
    for (int r = 0; r < 8; ++r) s[r] = 0.0f;
    {
        size_t keybase = ((size_t)(bh * T_ + kt * 16 + l15)) * HD_;
        #pragma unroll
        for (int f = 0; f < 2; ++f) {
            Frag kf;
            int dbase = f * 32 + half * 8;
            kf.q[0] = *(const uint4v*)(kbp + keybase + dbase);
            kf.q[1] = *(const uint4v*)(kbp + keybase + dbase + 16);
            s = __builtin_amdgcn_wmma_f32_16x16x32_bf16(
                    false, qf[f].v, false, kf.v, (short)0, s, false, false);
        }
    }

    // Online softmax update (f32). Row i lives on 16 lanes of one half-group.
    const float scale = 0.125f;   // 1/sqrt(64)
    float a_[8];
    #pragma unroll
    for (int r = 0; r < 8; ++r) {
        int   i  = qt * 16 + r + 8 * half;
        int   j  = kt * 16 + l15;
        float sv = s[r] * scale;
        bool  ok = (j <= i) && (((i - j) < WIN_) || (j < SINK_));
        sv = ok ? sv : -1e30f;
        float rm = sv;
        rm = fmaxf(rm, __shfl_xor(rm, 1, 32));
        rm = fmaxf(rm, __shfl_xor(rm, 2, 32));
        rm = fmaxf(rm, __shfl_xor(rm, 4, 32));
        rm = fmaxf(rm, __shfl_xor(rm, 8, 32));
        float mnew = fmaxf(mrow[r], rm);
        float al   = __expf(mrow[r] - mnew);
        mrow[r]    = mnew;
        float p    = __expf(sv - mnew);
        float rs   = p;
        rs += __shfl_xor(rs, 1, 32);
        rs += __shfl_xor(rs, 2, 32);
        rs += __shfl_xor(rs, 4, 32);
        rs += __shfl_xor(rs, 8, 32);
        lrow[r] = lrow[r] * al + rs;
        a_[r]   = al;
        PsW[r + 8 * half][l15] = bf_bits(p);   // re-layout P via LDS
    }
    #pragma unroll
    for (int dt = 0; dt < 4; ++dt)
        #pragma unroll
        for (int r = 0; r < 8; ++r) o[dt][r] *= a_[r];

    asm volatile("s_wait_dscnt 0x0" ::: "memory");  // intra-wave LDS RAW fence

    // P as A-fragment, K padded to 32 with zeros (k>=16 contributes 0)
    Frag pf;
    {
        int kb8 = half * 8;
        pf.q[0] = *(const uint4v*)&PsW[l15][kb8];
        #pragma unroll
        for (int i = 8; i < 16; ++i) pf.u[i] = 0;
    }
    #pragma unroll
    for (int dt = 0; dt < 4; ++dt) {
        Frag vf;
        int dcol = dt * 16 + l15;
        int kb16 = half * 8;
        vf.q[0] = *(const uint4v*)&VsW[dcol][kb16];
        vf.q[1] = *(const uint4v*)&VsW[dcol][kb16 + 16];
        o[dt] = __builtin_amdgcn_wmma_f32_16x16x32_bf16(
                    false, pf.v, false, vf.v, (short)0, o[dt], false, false);
    }
}

__global__ __launch_bounds__(128) void attn_swa(const unsigned short* __restrict__ qb,
                                                const unsigned short* __restrict__ kb,
                                                const unsigned short* __restrict__ vb,
                                                unsigned short* __restrict__ yb) {
    __shared__ __align__(16) unsigned short Ps[4][16][16];
    __shared__ __align__(16) unsigned short Vs[4][64][32];

    const int tid  = threadIdx.x;
    const int lane = tid & 31;
    const int w    = tid >> 5;
    const int half = lane >> 4;
    const int l15  = lane & 15;

    int g  = blockIdx.x * 4 + w;        // 0..4095 query tiles
    int bh = g >> 7;                    // /128 tiles per (b,h)
    int qt = g & 127;
    int b  = bh >> 4, h = bh & 15;

    unsigned short (*PsW)[16] = Ps[w];
    unsigned short (*VsW)[32] = Vs[w];

    // Zero the K-padding region of Vs once (k = 16..31 of each d row)
    for (int z = lane; z < 64 * 16; z += 32) {
        int dz = z >> 4, kz = z & 15;
        VsW[dz][16 + kz] = 0;
    }

    // Load Q fragments directly from global (A-layout: row=l15, K split by half)
    Frag qf[2];
    {
        size_t rowbase = ((size_t)(bh * T_ + qt * 16 + l15)) * HD_;
        int kb8 = half * 8;
        #pragma unroll
        for (int f = 0; f < 2; ++f) {
            qf[f].q[0] = *(const uint4v*)(qb + rowbase + f * 32 + kb8);
            qf[f].q[1] = *(const uint4v*)(qb + rowbase + f * 32 + kb8 + 16);
        }
    }

    v8f   o[4];
    float mrow[8], lrow[8];
    #pragma unroll
    for (int dt = 0; dt < 4; ++dt)
        #pragma unroll
        for (int r = 0; r < 8; ++r) o[dt][r] = 0.0f;
    #pragma unroll
    for (int r = 0; r < 8; ++r) { mrow[r] = -1e30f; lrow[r] = 0.0f; }

    int ktlo = qt - 16; if (ktlo < 0) ktlo = 0;   // WINDOW=256 => 16 key tiles
    if (ktlo > 0)   // sink tile (keys 0..15, mask keeps j < SINK)
        attn_tile(lane, bh, qt, 0, qf, kb, vb, o, mrow, lrow, PsW, VsW);
    for (int kt = ktlo; kt <= qt; ++kt)
        attn_tile(lane, bh, qt, kt, qf, kb, vb, o, mrow, lrow, PsW, VsW);

    // Normalize and write y[b, t, h*64+d] as bf16 (feeds the final GEMM)
    #pragma unroll
    for (int r = 0; r < 8; ++r) {
        float inv  = 1.0f / lrow[r];
        int   trow = qt * 16 + r + 8 * half;
        size_t ybase = ((size_t)(b * T_ + trow)) * C_ + h * HD_;
        #pragma unroll
        for (int dt = 0; dt < 4; ++dt)
            yb[ybase + dt * 16 + l15] = bf_bits(o[dt][r] * inv);
    }
}

// ---------------------------------------------------------------------------
extern "C" void kernel_launch(void* const* d_in, const int* in_sizes, int n_in,
                              void* d_out, int out_size, void* d_ws, size_t ws_size,
                              hipStream_t stream) {
    (void)in_sizes; (void)n_in; (void)out_size; (void)ws_size;
    const float* x      = (const float*)d_in[0];
    const float* w_attn = (const float*)d_in[1];
    const float* w_proj = (const float*)d_in[2];
    float*       out    = (float*)d_out;

    char* ws = (char*)d_ws;
    float*          qkv = (float*)ws;                           // 50,331,648 B
    unsigned short* qb  = (unsigned short*)(ws + 50331648);     //  8,388,608 B
    unsigned short* kb  = (unsigned short*)(ws + 58720256);     //  8,388,608 B
    unsigned short* vb  = (unsigned short*)(ws + 67108864);     //  8,388,608 B
    unsigned short* yb  = (unsigned short*)(ws + 75497472);     //  8,388,608 B
    unsigned short* xb  = (unsigned short*)(ws + 83886080);     //  8,388,608 B
    unsigned short* wab = (unsigned short*)(ws + 92274688);     //  6,291,456 B
    unsigned short* wpb = (unsigned short*)(ws + 98566144);     //  2,097,152 B

    // 0) one-time f32 -> bf16 conversions (halves GEMM read traffic)
    cvt_bf16<<<4096, 256, 0, stream>>>(x,      xb,  B_ * T_ * C_);
    cvt_bf16<<<3072, 256, 0, stream>>>(w_attn, wab, C_ * 3 * C_);
    cvt_bf16<<<1024, 256, 0, stream>>>(w_proj, wpb, C_ * C_);

    // 1) qkv = x @ w_attn   (M=4096, N=3072, K=1024)
    gemm_bf16<<<dim3(24, 32), 256, 0, stream>>>(xb, wab, qkv, 1024, 3072);
    // 2) RoPE + split to bf16 head-major layout
    rope_split<<<16384, 256, 0, stream>>>(qkv, qb, kb, vb);
    // 3) sliding-window flash attention
    attn_swa<<<1024, 128, 0, stream>>>(qb, kb, vb, yb);
    // 4) out = y @ w_proj   (M=4096, N=1024, K=1024)
    gemm_bf16<<<dim3(8, 32), 256, 0, stream>>>(yb, wpb, out, 1024, 1024);
}
